// GraphUNetv2_21784074126014
// MI455X (gfx1250) — matrix operand order repaired and checked by reading, not compiled, not run on hardware
//
#include <hip/hip_runtime.h>
#include <cstdint>

// ---------------------------------------------------------------------------
// Types for CDNA5 WMMA (wave32)
// ---------------------------------------------------------------------------
typedef __attribute__((ext_vector_type(16))) __bf16 v16bf;
typedef __attribute__((ext_vector_type(8)))  float  v8f;
typedef __attribute__((ext_vector_type(4)))  unsigned int v4u;
typedef __attribute__((ext_vector_type(8)))  int    v8i;
typedef __attribute__((ext_vector_type(4)))  int    v4i;

union Frag {
    v16bf          bf;
    unsigned int   u[8];
    unsigned short h[16];
};

__device__ __forceinline__ unsigned short f2bf(float f) {
    unsigned int u = __builtin_bit_cast(unsigned int, f);
    u += 0x7fffu + ((u >> 16) & 1u);          // round-to-nearest-even
    return (unsigned short)(u >> 16);
}

// ---------------------------------------------------------------------------
// Tensor Data Mover (gfx1250). Descriptor layout per CDNA5 ISA ch.8.
// 5-arg builtin on ROCm 7.2, 6-arg on the therock/clang-23 lane (which ships
// the gfx1250 TDM header) -- discriminate with __has_include.
// ---------------------------------------------------------------------------
#if defined(__HIP_DEVICE_COMPILE__) && \
    __has_builtin(__builtin_amdgcn_tensor_load_to_lds) && \
    __has_builtin(__builtin_amdgcn_s_wait_tensorcnt)
#define USE_TDM 1
#if __has_include(<hip/amd_detail/amd_gfx1250_TDM.h>)
#define TDM_ISSUE(g0, g1) \
    __builtin_amdgcn_tensor_load_to_lds(g0, g1, (v4i)0, (v4i)0, (v8i)0, 0)
#else
#define TDM_ISSUE(g0, g1) \
    __builtin_amdgcn_tensor_load_to_lds(g0, g1, (v4i)0, (v4i)0, 0)
#endif
#else
#define USE_TDM 0
#endif

#if USE_TDM
// 2D bf16 tile load: tile 32 x 128 (K x M), tensor row stride = K elements.
// pad: insert 4 dwords after every 16 dwords -> LDS row stride 20 dwords,
// matching the bank-conflict-free fragment-read stride below.
__device__ __forceinline__ void tdm_load_a(const unsigned short* g,
                                           unsigned int lds_off, int K, int M) {
    unsigned long long ga = (unsigned long long)(uintptr_t)g;
    v4u g0;
    g0[0] = 1u;                                       // count=1, user descriptor
    g0[1] = lds_off;                                  // lds_addr (bytes)
    g0[2] = (unsigned int)ga;                         // global_addr[31:0]
    g0[3] = (unsigned int)((ga >> 32) & 0x01ffffffull) | (2u << 30); // [56:32]+type=2
    v8i g1;
    g1[0] = (int)((1u << 16)                          // data_size = 2 bytes
                | (1u << 20)                          // pad_enable
                | (3u << 22)                          // pad_interval = 16 dwords
                | (3u << 25));                        // pad_amount = 4 dwords
    g1[1] = (int)(((unsigned)K & 0xffffu) << 16);     // tensor_dim0 lo
    g1[2] = (int)((((unsigned)K >> 16) & 0xffffu) |   // tensor_dim0 hi
                  (((unsigned)M & 0xffffu) << 16));   // tensor_dim1 lo
    g1[3] = (int)((((unsigned)M >> 16) & 0xffffu) |   // tensor_dim1 hi
                  (32u << 16));                       // tile_dim0 = 32
    g1[4] = 128;                                      // tile_dim1 = 128, tile_dim2 = 0
    g1[5] = K;                                        // tensor_dim0_stride[31:0]
    g1[6] = 0;
    g1[7] = 0;
    TDM_ISSUE(g0, g1);
}
#endif

// ---------------------------------------------------------------------------
// Generic bf16 GEMM:  C[M,N] = A[M,K] @ B[K,N]  (+bias) (+relu) (+zero_diag)
// A,B row-major bf16 (u16 bits), C f32. M%128==0, N%64==0, K%64==0.
// 256 threads = 8 waves. Block tile 128x64, K-step 32, double-buffered LDS
// with the pipeline manually unrolled x2 so buffer indices are compile-time
// constants (keeps fragment reads on the DS path, not flat).
// ---------------------------------------------------------------------------
#define GEMM_RELU 1
#define GEMM_ZERODIAG 2

__global__ __launch_bounds__(256)
void gemm_bf16_wmma(const unsigned short* __restrict__ A,
                    const unsigned short* __restrict__ B,
                    float* __restrict__ C,
                    const float* __restrict__ bias,
                    int M, int N, int K, int flags)
{
    __shared__ unsigned short lA[2][128 * 40];   // [m][k], stride 40 u16 = 20 dw
    __shared__ unsigned short lB[2][64 * 40];    // [n][k] transposed, same stride

    const int tid  = threadIdx.x;
    const int wave = tid >> 5;
    const int lane = tid & 31;
    const int half = lane >> 4;
    const int l16  = lane & 15;
    const int bm   = blockIdx.y * 128;
    const int bn   = blockIdx.x * 64;
    const int rowA = wave * 16 + l16;

    // ---- B staging coords: thread -> (row br, 8 cols at bc4*8)
    const int br  = tid >> 3;                     // 0..31
    const int bc4 = tid & 7;                      // 0..7
    const uint4* bg = (const uint4*)(B + (size_t)br * N + bn + bc4 * 8);
    const size_t bAdv = (size_t)4 * N;            // 32 rows in uint4 units

    auto stB = [&](int b, uint4 v) {
        unsigned int w[4] = {v.x, v.y, v.z, v.w};
        #pragma unroll
        for (int q = 0; q < 8; ++q)
            lB[b][(bc4 * 8 + q) * 40 + br] =
                (unsigned short)((w[q >> 1] >> ((q & 1) * 16)) & 0xffffu);
    };

#if USE_TDM
    const unsigned int ldsA[2] = {
        (unsigned int)(unsigned long long)(uintptr_t)&lA[0][0],
        (unsigned int)(unsigned long long)(uintptr_t)&lA[1][0]
    };
#else
    // ---- fallback A staging: 2 x uint4 per thread, batched then stored b128
    const int ar0  = tid >> 2;                    // 0..63 (second half +64)
    const int ac4  = tid & 3;                     // 0..3  (8 bf16 each)
    const uint4* ag0 = (const uint4*)(A + (size_t)(bm + ar0) * K + ac4 * 8);
    const uint4* ag1 = (const uint4*)(A + (size_t)(bm + ar0 + 64) * K + ac4 * 8);
    uint4 nva0, nva1;
#endif

#if USE_TDM
#define STAGE_A_ISSUE(NXT, KNEXT) \
    if (wave == 0) tdm_load_a(A + (size_t)bm * K + (KNEXT), ldsA[NXT], K, M);
#define STAGE_A_COMMIT(NXT) \
    if (wave == 0) __builtin_amdgcn_s_wait_tensorcnt(0);
#else
#define STAGE_A_ISSUE(NXT, KNEXT) \
    { nva0 = *ag0; nva1 = *ag1; ag0 += 4; ag1 += 4; }
#define STAGE_A_COMMIT(NXT) \
    { *(uint4*)&lA[NXT][ar0 * 40 + ac4 * 8]        = nva0; \
      *(uint4*)&lA[NXT][(ar0 + 64) * 40 + ac4 * 8] = nva1; }
#endif

    // ---- prologue: stage tile 0 into buffer 0
#if USE_TDM
    if (wave == 0) tdm_load_a(A + (size_t)bm * K, ldsA[0], K, M);
#else
    STAGE_A_ISSUE(0, 0)
    STAGE_A_COMMIT(0)
#endif
    {
        uint4 vb = *bg; bg += bAdv;
        stB(0, vb);
    }
#if USE_TDM
    if (wave == 0) __builtin_amdgcn_s_wait_tensorcnt(0);
#endif
    __syncthreads();

    v8f acc[4] = {};

    // One pipeline phase: compute buffer CUR, prefetch tile KNEXT into NXT.
    // CUR/NXT are textual literals so LDS indexing stays address-space-folded.
#define PHASE(CUR, NXT, HAS_NEXT, KNEXT)                                        \
    {                                                                           \
        uint4 nvb;                                                              \
        if (HAS_NEXT) {                                                         \
            STAGE_A_ISSUE(NXT, KNEXT)                                           \
            nvb = *bg; bg += bAdv;                                              \
        }                                                                       \
        Frag a;                                                                 \
        _Pragma("unroll")                                                       \
        for (int d = 0; d < 8; ++d) {                                           \
            int k = ((d & 4) << 2) + half * 8 + ((d & 3) << 1);                 \
            a.u[d] = *(const unsigned int*)&lA[CUR][rowA * 40 + k];             \
        }                                                                       \
        _Pragma("unroll")                                                       \
        for (int t = 0; t < 4; ++t) {                                           \
            Frag b;                                                             \
            _Pragma("unroll")                                                   \
            for (int d = 0; d < 8; ++d)                                         \
                b.u[d] = *(const unsigned int*)                                 \
                          &lB[CUR][(t * 16 + l16) * 40 + ((d & 4) << 2) + half * 8 + ((d & 3) << 1)];    \
            acc[t] = __builtin_amdgcn_wmma_f32_16x16x32_bf16(                   \
                         false, a.bf, false, b.bf, (short)0, acc[t],            \
                         false, false);                                         \
        }                                                                       \
        if (HAS_NEXT) {                                                         \
            STAGE_A_COMMIT(NXT)                                                 \
            stB(NXT, nvb);                                                      \
        }                                                                       \
        __syncthreads();                                                        \
    }
// Note: in the TDM path STAGE_A_COMMIT is the tensorcnt wait (must precede the
// barrier); in the fallback it is the LDS store of the staged registers.

    int k0 = 0;
    for (; k0 + 64 < K; k0 += 64) {
        PHASE(0, 1, true, k0 + 32)
        PHASE(1, 0, true, k0 + 64)
    }
    PHASE(0, 1, true, k0 + 32)      // K % 64 == 0 -> k0 + 32 always valid
    PHASE(1, 0, false, 0)
#undef PHASE
#undef STAGE_A_ISSUE
#undef STAGE_A_COMMIT

    // ---- epilogue: C layout vgpr v -> row = v + half*8, col = l16
    #pragma unroll
    for (int t = 0; t < 4; ++t) {
        int col = bn + t * 16 + l16;
        float bv = bias ? bias[col] : 0.0f;
        #pragma unroll
        for (int v = 0; v < 8; ++v) {
            int row = bm + wave * 16 + half * 8 + v;
            float x = acc[t][v] + bv;
            if (flags & GEMM_RELU) x = fmaxf(x, 0.0f);
            if ((flags & GEMM_ZERODIAG) && row == col) x = 0.0f;
            C[(size_t)row * N + col] = x;
        }
    }
}

// ---------------------------------------------------------------------------
// Support kernels
// ---------------------------------------------------------------------------
__global__ void k_scatter_edges(float* __restrict__ A, const int* __restrict__ ei,
                                int E, int N) {
    int e = blockIdx.x * blockDim.x + threadIdx.x;
    if (e < E) atomicAdd(&A[(size_t)ei[e] * N + ei[E + e]], 1.0f);
}

__global__ __launch_bounds__(256)
void k_row_dinv(const float* __restrict__ A, int n, float* __restrict__ dinv) {
    __shared__ float red[256];
    int r = blockIdx.x;
    float s = 0.f;
    for (int c = threadIdx.x; c < n; c += 256) s += A[(size_t)r * n + c];
    red[threadIdx.x] = s; __syncthreads();
    for (int w = 128; w > 0; w >>= 1) {
        if (threadIdx.x < w) red[threadIdx.x] += red[threadIdx.x + w];
        __syncthreads();
    }
    if (threadIdx.x == 0) dinv[r] = rsqrtf(red[0] + 1.0f);   // deg of (A+I)
}

__global__ void k_make_ahat(const float* __restrict__ A, const float* __restrict__ dinv,
                            int n, unsigned short* __restrict__ out) {
    size_t i = (size_t)blockIdx.x * blockDim.x + threadIdx.x;
    if (i >= (size_t)n * n) return;
    int r = (int)(i / n), c = (int)(i % n);
    float v = A[i] + (r == c ? 1.0f : 0.0f);
    out[i] = f2bf(dinv[r] * v * dinv[c]);
}

__global__ void k_make_aug(const float* __restrict__ A, int n,
                           unsigned short* __restrict__ out) {
    size_t i = (size_t)blockIdx.x * blockDim.x + threadIdx.x;
    if (i >= (size_t)n * n) return;
    int r = (int)(i / n), c = (int)(i % n);
    out[i] = f2bf(r == c ? 1.0f : A[i]);   // zero_diag(A) + I
}

__global__ void k_cvt_bf16(const float* __restrict__ s, unsigned short* __restrict__ d,
                           size_t n) {
    size_t i = (size_t)blockIdx.x * blockDim.x + threadIdx.x;
    if (i < n) d[i] = f2bf(s[i]);
}

__global__ void k_inv_norm(const float* __restrict__ w, int c, float* __restrict__ out) {
    if (threadIdx.x == 0 && blockIdx.x == 0) {
        float s = 0.f;
        for (int i = 0; i < c; ++i) s += w[i] * w[i];
        *out = rsqrtf(s);
    }
}

__global__ void k_score(const float* __restrict__ x, const float* __restrict__ w,
                        const float* __restrict__ invn, int n, int c,
                        float* __restrict__ score) {
    int i = blockIdx.x * blockDim.x + threadIdx.x;
    if (i >= n) return;
    float s = 0.f;
    for (int j = 0; j < c; ++j) s += x[(size_t)i * c + j] * w[j];
    score[i] = tanhf(s * (*invn));
}

// Full descending bitonic sort of n<=4096 scores in one workgroup; emit top-k.
__global__ __launch_bounds__(1024)
void k_topk(const float* __restrict__ score, int n, int k,
            int* __restrict__ perm, float* __restrict__ vals) {
    __shared__ float skey[4096];
    __shared__ int   sidx[4096];
    int tid = threadIdx.x;
    for (int i = tid; i < n; i += 1024) { skey[i] = score[i]; sidx[i] = i; }
    __syncthreads();
    for (int kk = 2; kk <= n; kk <<= 1) {
        for (int j = kk >> 1; j > 0; j >>= 1) {
            for (int i = tid; i < n; i += 1024) {
                int ij = i ^ j;
                if (ij > i) {
                    bool descend = ((i & kk) == 0);
                    float ka = skey[i], kb = skey[ij];
                    int   ia = sidx[i], ib = sidx[ij];
                    bool aBelow = (ka < kb) || (ka == kb && ia > ib);
                    if (aBelow == descend) {
                        skey[i] = kb; skey[ij] = ka;
                        sidx[i] = ib; sidx[ij] = ia;
                    }
                }
            }
            __syncthreads();
        }
    }
    for (int i = tid; i < k; i += 1024) { perm[i] = sidx[i]; vals[i] = skey[i]; }
}

__global__ void k_gather_scale(const float* __restrict__ x, const int* __restrict__ perm,
                               const float* __restrict__ vals, int k, int c,
                               float* __restrict__ out) {
    int i = blockIdx.x * blockDim.x + threadIdx.x;
    if (i >= k * c) return;
    int r = i / c, col = i % c;
    out[i] = x[(size_t)perm[r] * c + col] * vals[r];
}

__global__ void k_gather_adj(const float* __restrict__ A2, int n,
                             const int* __restrict__ perm, int k,
                             float* __restrict__ out) {
    size_t i = (size_t)blockIdx.x * blockDim.x + threadIdx.x;
    if (i >= (size_t)k * k) return;
    int r = (int)(i / k), c = (int)(i % k);
    out[i] = A2[(size_t)perm[r] * n + perm[c]];
}

__global__ void k_scatter_add(float* __restrict__ out, const float* __restrict__ x,
                              const int* __restrict__ perm, int k, int c) {
    int i = blockIdx.x * blockDim.x + threadIdx.x;
    if (i >= k * c) return;
    int r = i / c, col = i % c;
    out[(size_t)perm[r] * c + col] += x[i];   // perm unique -> set == add
}

// ---------------------------------------------------------------------------
// Host orchestration
// ---------------------------------------------------------------------------
static void gemm(const unsigned short* A, const unsigned short* B, float* C,
                 const float* bias, int M, int N, int K, int flags, hipStream_t s) {
    dim3 g(N / 64, M / 128);
    gemm_bf16_wmma<<<g, 256, 0, s>>>(A, B, C, bias, M, N, K, flags);
}
static void cvt(const float* s, unsigned short* d, size_t n, hipStream_t st) {
    k_cvt_bf16<<<(unsigned)((n + 255) / 256), 256, 0, st>>>(s, d, n);
}

extern "C" void kernel_launch(void* const* d_in, const int* in_sizes, int n_in,
                              void* d_out, int out_size, void* d_ws, size_t ws_size,
                              hipStream_t stream) {
    (void)in_sizes; (void)n_in; (void)out_size; (void)ws_size;
    const int N0 = 4096, N1 = 2048, N2 = 1024, HID = 128, INC = 64, OUTC = 64;
    const int E = 65536;

    const float* x_in = (const float*)d_in[0];
    const int*   ei   = (const int*)d_in[1];
    const float* W[6]  = {(const float*)d_in[2],  (const float*)d_in[4],
                          (const float*)d_in[6],  (const float*)d_in[8],
                          (const float*)d_in[10], (const float*)d_in[12]};
    const float* Bv[6] = {(const float*)d_in[3],  (const float*)d_in[5],
                          (const float*)d_in[7],  (const float*)d_in[9],
                          (const float*)d_in[11], (const float*)d_in[13]};
    const float* pw0 = (const float*)d_in[14];
    const float* pw1 = (const float*)d_in[15];
    const float* Wu[4]  = {(const float*)d_in[16], (const float*)d_in[18],
                           (const float*)d_in[20], (const float*)d_in[22]};
    const float* Bu[4]  = {(const float*)d_in[17], (const float*)d_in[19],
                           (const float*)d_in[21], (const float*)d_in[23]};

    // workspace layout
    size_t off = 0;
    auto alloc = [&](size_t bytes) -> char* {
        char* p = (char*)d_ws + off;
        off += (bytes + 255) & ~(size_t)255;
        return p;
    };
    float* bigA  = (float*)alloc((size_t)N0 * N0 * 4);         // A0, later A0^2
    unsigned short* Ahat0 = (unsigned short*)alloc((size_t)N0 * N0 * 2);
    unsigned short* Aug0  = (unsigned short*)alloc((size_t)N0 * N0 * 2);
    float* adj1  = (float*)alloc((size_t)N1 * N1 * 4);         // adj1, later adj1^2
    unsigned short* Ahat1 = (unsigned short*)alloc((size_t)N1 * N1 * 2);
    unsigned short* Aug1  = (unsigned short*)alloc((size_t)N1 * N1 * 2);
    float* adj2  = (float*)alloc((size_t)N2 * N2 * 4);
    unsigned short* Ahat2 = (unsigned short*)alloc((size_t)N2 * N2 * 2);
    float* dinv0 = (float*)alloc(N0 * 4);
    float* dinv1 = (float*)alloc(N1 * 4);
    float* dinv2 = (float*)alloc(N2 * 4);
    float* score = (float*)alloc(N0 * 4);
    float* vals  = (float*)alloc(N0 * 4);
    int*   perm1 = (int*)alloc(N1 * 4);
    int*   perm2 = (int*)alloc(N2 * 4);
    float* invn  = (float*)alloc(256);
    float* xs0   = (float*)alloc((size_t)N0 * HID * 4);
    float* xs1   = (float*)alloc((size_t)N1 * HID * 4);
    float* fA    = (float*)alloc((size_t)N0 * HID * 4);
    float* fB    = (float*)alloc((size_t)N0 * HID * 4);
    unsigned short* xbf = (unsigned short*)alloc((size_t)N0 * HID * 2);
    unsigned short* tbf = (unsigned short*)alloc((size_t)N0 * HID * 2);
    unsigned short* Wb[6]; unsigned short* Wub[4];
    Wb[0] = (unsigned short*)alloc(INC * HID * 2);
    for (int i = 1; i < 6; ++i) Wb[i] = (unsigned short*)alloc(HID * HID * 2);
    for (int i = 0; i < 3; ++i) Wub[i] = (unsigned short*)alloc(HID * HID * 2);
    Wub[3] = (unsigned short*)alloc(HID * OUTC * 2);

    // weights -> bf16
    cvt(W[0], Wb[0], (size_t)INC * HID, stream);
    for (int i = 1; i < 6; ++i) cvt(W[i], Wb[i], (size_t)HID * HID, stream);
    for (int i = 0; i < 3; ++i) cvt(Wu[i], Wub[i], (size_t)HID * HID, stream);
    cvt(Wu[3], Wub[3], (size_t)HID * OUTC, stream);

    // ---- level 0 adjacency
    hipMemsetAsync(bigA, 0, (size_t)N0 * N0 * 4, stream);
    k_scatter_edges<<<E / 256, 256, 0, stream>>>(bigA, ei, E, N0);
    k_row_dinv<<<N0, 256, 0, stream>>>(bigA, N0, dinv0);
    {
        unsigned nb = (unsigned)(((size_t)N0 * N0 + 255) / 256);
        k_make_ahat<<<nb, 256, 0, stream>>>(bigA, dinv0, N0, Ahat0);
        k_make_aug<<<nb, 256, 0, stream>>>(bigA, N0, Aug0);
    }

    // ---- down block 0  (x: 4096x64 -> xs0: 4096x128)
    cvt(x_in, xbf, (size_t)N0 * INC, stream);
    gemm(xbf, Wb[0], fA, nullptr, N0, HID, INC, 0, stream);
    cvt(fA, tbf, (size_t)N0 * HID, stream);
    gemm(Ahat0, tbf, fB, Bv[0], N0, HID, N0, GEMM_RELU, stream);
    cvt(fB, xbf, (size_t)N0 * HID, stream);
    gemm(xbf, Wb[1], fA, nullptr, N0, HID, HID, 0, stream);
    cvt(fA, tbf, (size_t)N0 * HID, stream);
    gemm(Ahat0, tbf, xs0, Bv[1], N0, HID, N0, GEMM_RELU, stream);

    // ---- pool 1 (4096 -> 2048)
    k_inv_norm<<<1, 32, 0, stream>>>(pw0, HID, invn);
    k_score<<<N0 / 256, 256, 0, stream>>>(xs0, pw0, invn, N0, HID, score);
    k_topk<<<1, 1024, 0, stream>>>(score, N0, N1, perm1, vals);
    k_gather_scale<<<(N1 * HID) / 256, 256, 0, stream>>>(xs0, perm1, vals, N1, HID, fA);
    gemm(Aug0, Aug0, bigA, nullptr, N0, N0, N0, GEMM_ZERODIAG, stream);   // A^2
    k_gather_adj<<<(unsigned)(((size_t)N1 * N1 + 255) / 256), 256, 0, stream>>>(
        bigA, N0, perm1, N1, adj1);
    k_row_dinv<<<N1, 256, 0, stream>>>(adj1, N1, dinv1);
    {
        unsigned nb = (unsigned)(((size_t)N1 * N1 + 255) / 256);
        k_make_ahat<<<nb, 256, 0, stream>>>(adj1, dinv1, N1, Ahat1);
        k_make_aug<<<nb, 256, 0, stream>>>(adj1, N1, Aug1);
    }

    // ---- down block 1 (fA: 2048x128 -> xs1)
    cvt(fA, xbf, (size_t)N1 * HID, stream);
    gemm(xbf, Wb[2], fB, nullptr, N1, HID, HID, 0, stream);
    cvt(fB, tbf, (size_t)N1 * HID, stream);
    gemm(Ahat1, tbf, fA, Bv[2], N1, HID, N1, GEMM_RELU, stream);
    cvt(fA, xbf, (size_t)N1 * HID, stream);
    gemm(xbf, Wb[3], fB, nullptr, N1, HID, HID, 0, stream);
    cvt(fB, tbf, (size_t)N1 * HID, stream);
    gemm(Ahat1, tbf, xs1, Bv[3], N1, HID, N1, GEMM_RELU, stream);

    // ---- pool 2 (2048 -> 1024)
    k_inv_norm<<<1, 32, 0, stream>>>(pw1, HID, invn);
    k_score<<<N1 / 256, 256, 0, stream>>>(xs1, pw1, invn, N1, HID, score);
    k_topk<<<1, 1024, 0, stream>>>(score, N1, N2, perm2, vals);
    k_gather_scale<<<(N2 * HID) / 256, 256, 0, stream>>>(xs1, perm2, vals, N2, HID, fA);
    gemm(Aug1, Aug1, adj1, nullptr, N1, N1, N1, GEMM_ZERODIAG, stream);   // reuse adj1 buf
    k_gather_adj<<<(unsigned)(((size_t)N2 * N2 + 255) / 256), 256, 0, stream>>>(
        adj1, N1, perm2, N2, adj2);
    k_row_dinv<<<N2, 256, 0, stream>>>(adj2, N2, dinv2);
    k_make_ahat<<<(unsigned)(((size_t)N2 * N2 + 255) / 256), 256, 0, stream>>>(
        adj2, dinv2, N2, Ahat2);

    // ---- down block 2 (fA: 1024x128)
    cvt(fA, xbf, (size_t)N2 * HID, stream);
    gemm(xbf, Wb[4], fB, nullptr, N2, HID, HID, 0, stream);
    cvt(fB, tbf, (size_t)N2 * HID, stream);
    gemm(Ahat2, tbf, fA, Bv[4], N2, HID, N2, GEMM_RELU, stream);
    cvt(fA, xbf, (size_t)N2 * HID, stream);
    gemm(xbf, Wb[5], fB, nullptr, N2, HID, HID, 0, stream);
    cvt(fB, tbf, (size_t)N2 * HID, stream);
    gemm(Ahat2, tbf, fA, Bv[5], N2, HID, N2, GEMM_RELU, stream);   // x @ depth

    // ---- up block 0 (res = xs1, adj = Ahat1, perm = perm2)
    hipMemcpyAsync(fB, xs1, (size_t)N1 * HID * 4, hipMemcpyDeviceToDevice, stream);
    k_scatter_add<<<(N2 * HID) / 256, 256, 0, stream>>>(fB, fA, perm2, N2, HID);
    cvt(fB, xbf, (size_t)N1 * HID, stream);
    gemm(xbf, Wub[0], fA, nullptr, N1, HID, HID, 0, stream);
    cvt(fA, tbf, (size_t)N1 * HID, stream);
    gemm(Ahat1, tbf, fB, Bu[0], N1, HID, N1, GEMM_RELU, stream);
    cvt(fB, xbf, (size_t)N1 * HID, stream);
    gemm(xbf, Wub[1], fA, nullptr, N1, HID, HID, 0, stream);
    cvt(fA, tbf, (size_t)N1 * HID, stream);
    gemm(Ahat1, tbf, fB, Bu[1], N1, HID, N1, GEMM_RELU, stream);   // outer relu (i<DEPTH-1)

    // ---- up block 1 (res = xs0, adj = Ahat0, perm = perm1)
    hipMemcpyAsync(fA, xs0, (size_t)N0 * HID * 4, hipMemcpyDeviceToDevice, stream);
    k_scatter_add<<<(N1 * HID) / 256, 256, 0, stream>>>(fA, fB, perm1, N1, HID);
    cvt(fA, xbf, (size_t)N0 * HID, stream);
    gemm(xbf, Wub[2], fB, nullptr, N0, HID, HID, 0, stream);
    cvt(fB, tbf, (size_t)N0 * HID, stream);
    gemm(Ahat0, tbf, fA, Bu[2], N0, HID, N0, GEMM_RELU, stream);   // inner relu
    cvt(fA, xbf, (size_t)N0 * HID, stream);
    gemm(xbf, Wub[3], fB, nullptr, N0, OUTC, HID, 0, stream);
    cvt(fB, tbf, (size_t)N0 * OUTC, stream);
    gemm(Ahat0, tbf, (float*)d_out, Bu[3], N0, OUTC, N0, 0, stream);  // final, no relu
}
